// VectorQuantizer_74423193305763
// MI455X (gfx1250) — matrix operand order repaired and checked by reading, not compiled, not run on hardware
//
#include <hip/hip_runtime.h>
#include <hip/hip_bf16.h>

typedef __attribute__((ext_vector_type(16))) _Float16 v16h;
typedef __attribute__((ext_vector_type(8)))  _Float16 v8h;
typedef __attribute__((ext_vector_type(8)))  float    v8f;
typedef __attribute__((ext_vector_type(2)))  float    v2f;
typedef __attribute__((ext_vector_type(4)))  unsigned int v4u;
typedef __attribute__((ext_vector_type(8)))  int      v8i;
typedef __attribute__((ext_vector_type(4)))  int      v4i;

#define N_E   8192
#define E_DIM 256
#define NROWS 16384              // 2 * 8*32*32
#define SPAT  8192               // 8*32*32
#define NELEM 4194304            // 2*256*8192
#define NCHUNK 4
#define CHUNK  (N_E / NCHUNK)    // 2048 codes per column chunk
#define TILE_N 32                // codes staged per TDM transfer
#define NTILE  (CHUNK / TILE_N)  // 64 tiles per chunk
#define BSTRIDE 264              // padded LDS row stride in halves (528 B)

// ---- workspace layout (bytes) ----
#define WS_SUMS   0              // float[4]
#define WS_HIST   4096           // int[8192]
#define WS_IDX    36864          // int[16384]
#define WS_CNORM  102400         // float[8192]
#define WS_PVAL   135168         // float[4*16384] partial min values
#define WS_PIDX   397312         // int[4*16384]   partial min indices
#define WS_ZH     659456         // f16[16384*256]
#define WS_CBH    9048064        // f16[8192*256]

// ---- output layout (floats) ----
#define OUT_ZQ    0
#define OUT_LOSS  4194304
#define OUT_PPL   4194305
#define OUT_ME    4194306
#define OUT_IDX   138412034      // 4194306 + 16384*8192

// -------- 0: zero the small accumulators (sums + hist) --------
__global__ void vq_init_kernel(int* wsi) {
    int i = blockIdx.x * 256 + threadIdx.x;
    if (i < 9216) wsi[i] = 0;    // covers sums @0 and hist @4096 bytes
}

// -------- 1: transpose z [2,256,8192] f32 -> zh [16384,256] f16 --------
__global__ __launch_bounds__(256) void vq_prepz_kernel(
        const float* __restrict__ z, _Float16* __restrict__ zh) {
    __shared__ float tile[32][33];
    int blk = blockIdx.x;
    int b   = blk >> 11;
    int rem = blk & 2047;
    int c0  = (rem >> 8) << 5;
    int s0  = (rem & 255) << 5;
    int tid = threadIdx.x;
    #pragma unroll
    for (int k = 0; k < 4; ++k) {
        int lin = k * 256 + tid;
        int cc = lin >> 5, ss = lin & 31;
        tile[cc][ss] = z[(size_t)b * 2097152 + (size_t)(c0 + cc) * SPAT + s0 + ss];
    }
    __syncthreads();
    #pragma unroll
    for (int k = 0; k < 4; ++k) {
        int lin = k * 256 + tid;
        int rr = lin >> 5, col = lin & 31;
        zh[(size_t)(b * SPAT + s0 + rr) * E_DIM + c0 + col] = (_Float16)tile[col][rr];
    }
}

// -------- 2: codebook f32 -> f16, and row norms ||e||^2 --------
__global__ __launch_bounds__(64) void vq_prepcb_kernel(
        const float* __restrict__ cb, _Float16* __restrict__ cbh,
        float* __restrict__ cnorm) {
    __shared__ float red[64];
    int row = blockIdx.x, t = threadIdx.x;
    float acc = 0.f;
    #pragma unroll
    for (int k = 0; k < 4; ++k) {
        int j = t + k * 64;
        float v = cb[(size_t)row * E_DIM + j];
        cbh[(size_t)row * E_DIM + j] = (_Float16)v;
        acc += v * v;
    }
    red[t] = acc;
    __syncthreads();
    for (int off = 32; off > 0; off >>= 1) {
        if (t < off) red[t] += red[t + off];
        __syncthreads();
    }
    if (t == 0) cnorm[row] = red[0];
}

// Issue one TDM copy: 32 codebook rows (32 x 256 f16 = 16 KB) -> LDS, with
// pad_enable=1, pad_interval=6 (128 DWORDs = 512B), pad_amount=3 (4 DWORDs =
// 16B) => LDS row stride 528B = 264 halves, matching the B-frag addressing.
__device__ __forceinline__ void tdm_issue_tile(const _Float16* gsrc,
                                               unsigned lds_off) {
    unsigned long long ga = (unsigned long long)(size_t)gsrc;
    v4u g0;
    g0.x = 1u;                                  // count=1, user mode
    g0.y = lds_off;                             // D#.lds_addr
    g0.z = (unsigned)ga;                        // global_addr[31:0]
    g0.w = (unsigned)(ga >> 32) | (2u << 30);   // global_addr[56:32] | type=2
    v8i g1;
    g1[0] = (1 << 16) | (1 << 20) | (6 << 22) | (3 << 25); // dsz=2B,pad cfg
    g1[1] = (E_DIM << 16);                      // tensor_dim0 = 256
    g1[2] = (N_E << 16);                        // tensor_dim1 = 8192
    g1[3] = (E_DIM << 16);                      // tile_dim0 = 256
    g1[4] = TILE_N;                             // tile_dim1 = 32
    g1[5] = E_DIM;                              // tensor_dim0_stride = 256
    g1[6] = 0;
    g1[7] = 0;
    v4i gz4 = {};
    v8i gz8 = {};
    __builtin_amdgcn_tensor_load_to_lds(g0, g1, gz4, gz4, gz8, 0);
}

// -------- 3: WMMA argmin over scores ||e||^2 - 2 z.e (per column chunk) ----
__global__ __launch_bounds__(128) void vq_argmin_kernel(
        const _Float16* __restrict__ zh, const _Float16* __restrict__ cbh,
        const float* __restrict__ cnorm, float* __restrict__ pval,
        int* __restrict__ pidx) {
    __shared__ _Float16 btile[2][TILE_N * BSTRIDE];  // double-buffered TDM dest
    __shared__ float rval[4][8][32];
    __shared__ int   ridx[4][8][32];

    const int tid   = threadIdx.x;
    const int wave  = tid >> 5;
    const int lane  = tid & 31;
    const int ln    = lane & 15;
    const int hh    = lane >> 4;               // lane half (0/1)
    const int rb    = blockIdx.x >> 2;         // row block 0..255
    const int chunk = blockIdx.x & 3;          // column chunk 0..3
    const int row0  = rb * 64 + wave * 16;
    const int colbase = chunk * CHUNK;

    // Preload all 8 A fragments (16 rows x K=256) into VGPRs, per ISA layout:
    // lane<16: K {k0..k0+7, k0+16..k0+23}; lane>=16: K {k0+8..15, k0+24..31}.
    v16h a[8];
    {
        const _Float16* zrow = zh + (size_t)(row0 + ln) * E_DIM;
        #pragma unroll
        for (int ks = 0; ks < 8; ++ks) {
            const _Float16* p = zrow + ks * 32 + hh * 8;
            v8h lo = *(const v8h*)(p);
            v8h hi = *(const v8h*)(p + 16);
            a[ks] = __builtin_shufflevector(lo, hi,
                     0,1,2,3,4,5,6,7,8,9,10,11,12,13,14,15);
        }
    }

    float minv[8]; int mini[8];
    #pragma unroll
    for (int i = 0; i < 8; ++i) { minv[i] = 3.4e38f; mini[i] = 0; }

    const unsigned lds0 = (unsigned)(size_t)(&btile[0][0]);
    const unsigned lds1 = (unsigned)(size_t)(&btile[1][0]);

    // Prologue: TDM-prefetch tiles 0 and 1 (wave 0 issues; one DMA per call).
    if (wave == 0) {
        tdm_issue_tile(cbh + (size_t)colbase * E_DIM, lds0);
        tdm_issue_tile(cbh + (size_t)(colbase + TILE_N) * E_DIM, lds1);
    }

    for (int t = 0; t < NTILE; ++t) {
        if (wave == 0) {
            if (t < NTILE - 1) __builtin_amdgcn_s_wait_tensorcnt(1);
            else               __builtin_amdgcn_s_wait_tensorcnt(0);
        }
        __syncthreads();                       // tile t resident for all waves

        // two 16-column sub-tiles per staged 32-column tile
        #pragma unroll
        for (int sub = 0; sub < 2; ++sub) {
            const _Float16* bp =
                &btile[t & 1][(sub * 16 + ln) * BSTRIDE + hh * 8];
            // preload ALL B fragments, then an uninterrupted WMMA burst
            v16h b[8];
            #pragma unroll
            for (int ks = 0; ks < 8; ++ks) {
                v8h lo = *(const v8h*)(bp + ks * 32);
                v8h hi = *(const v8h*)(bp + ks * 32 + 16);
                b[ks] = __builtin_shufflevector(lo, hi,
                         0,1,2,3,4,5,6,7,8,9,10,11,12,13,14,15);
            }
            v8f acc = {};
            #pragma unroll
            for (int ks = 0; ks < 8; ++ks) {
                acc = __builtin_amdgcn_wmma_f32_16x16x32_f16(
                          false, a[ks], false, b[ks], (short)0, acc,
                          false, false);
            }
            int col  = colbase + t * TILE_N + sub * 16 + ln;
            float cn = cnorm[col];
            #pragma unroll
            for (int i = 0; i < 8; ++i) {
                float s = fmaf(-2.0f, acc[i], cn);
                if (s < minv[i]) { minv[i] = s; mini[i] = col; }
            }
        }

        __syncthreads();                       // all reads of buf done
        if (wave == 0 && t + 2 < NTILE) {
            tdm_issue_tile(cbh + (size_t)(colbase + (t + 2) * TILE_N) * E_DIM,
                           (t & 1) ? lds1 : lds0);
        }
    }

    #pragma unroll
    for (int i = 0; i < 8; ++i) {
        rval[wave][i][lane] = minv[i];
        ridx[wave][i][lane] = mini[i];
    }
    __syncthreads();
    if (lane < 16) {
        // row r: r<8 -> vgpr r, lanes 0..15; r>=8 -> vgpr r-8, lanes 16..31
        int vi = lane & 7;
        int l0 = (lane >> 3) * 16;
        float best = 3.4e38f; int bi = 0;
        for (int l = 0; l < 16; ++l) {
            float v = rval[wave][vi][l0 + l];
            int  ix = ridx[wave][vi][l0 + l];
            if (v < best) { best = v; bi = ix; }
        }
        pval[chunk * NROWS + row0 + lane] = best;
        pidx[chunk * NROWS + row0 + lane] = bi;
    }
}

// -------- 3b: combine the 4 per-chunk partial argmins --------
__global__ __launch_bounds__(256) void vq_reduce_kernel(
        const float* __restrict__ pval, const int* __restrict__ pidx,
        int* __restrict__ idxbuf) {
    int n = blockIdx.x * 256 + threadIdx.x;
    float best = 3.4e38f; int bi = 0;
    #pragma unroll
    for (int c = 0; c < NCHUNK; ++c) {        // ascending -> first-min ties
        float v = pval[c * NROWS + n];
        int  ix = pidx[c * NROWS + n];
        if (v < best) { best = v; bi = ix; }
    }
    idxbuf[n] = bi;
}

// -------- 4: gather z_q, squared-error sum, histogram, idx-out --------
__global__ __launch_bounds__(256) void vq_gather_kernel(
        const float* __restrict__ z, const float* __restrict__ cb,
        const int* __restrict__ idxbuf, float* __restrict__ out_zq,
        float* __restrict__ sums, int* __restrict__ hist,
        float* __restrict__ idx_out) {
    __shared__ float red[256];
    int e = blockIdx.x * 256 + threadIdx.x;
    int s = e & (SPAT - 1);
    int c = (e >> 13) & 255;
    int b = e >> 21;
    int n = b * SPAT + s;
    int iv = idxbuf[n];
    float zq = cb[iv * E_DIM + c];
    float zt = z[e];
    float d = zq - zt;
    out_zq[e] = zq;
    if (c == 0) {
        atomicAdd(&hist[iv], 1);
        idx_out[n] = (float)iv;
    }
    red[threadIdx.x] = d * d;
    __syncthreads();
    for (int off = 128; off > 0; off >>= 1) {
        if (threadIdx.x < off) red[threadIdx.x] += red[threadIdx.x + off];
        __syncthreads();
    }
    if (threadIdx.x == 0) atomicAdd(sums, red[0]);
}

// -------- 5: one-hot encodings, 536 MB non-temporal stream --------
__global__ __launch_bounds__(256) void vq_onehot_kernel(
        const int* __restrict__ idxbuf, float* __restrict__ me) {
    const long long total = (long long)NROWS * (N_E / 2);
    long long stride = (long long)gridDim.x * blockDim.x;
    for (long long p = (long long)blockIdx.x * blockDim.x + threadIdx.x;
         p < total; p += stride) {
        int row = (int)(p >> 12);
        int j0  = ((int)p & 4095) * 2;
        int iv  = idxbuf[row];
        v2f v;
        v.x = (j0     == iv) ? 1.0f : 0.0f;
        v.y = (j0 + 1 == iv) ? 1.0f : 0.0f;
        __builtin_nontemporal_store(v, (v2f*)(me + (long long)row * N_E + j0));
    }
}

// -------- 6: loss + perplexity --------
__global__ __launch_bounds__(256) void vq_final_kernel(
        const float* __restrict__ sums, const int* __restrict__ hist,
        float* __restrict__ out) {
    __shared__ float red[256];
    float acc = 0.f;
    for (int j = threadIdx.x; j < N_E; j += 256) {
        float p = (float)hist[j] * (1.0f / (float)NROWS);
        acc += p * logf(p + 1e-10f);
    }
    red[threadIdx.x] = acc;
    __syncthreads();
    for (int off = 128; off > 0; off >>= 1) {
        if (threadIdx.x < off) red[threadIdx.x] += red[threadIdx.x + off];
        __syncthreads();
    }
    if (threadIdx.x == 0) {
        out[OUT_PPL]  = expf(-red[0]);
        out[OUT_LOSS] = 1.25f * sums[0] / (float)NELEM;
    }
}

extern "C" void kernel_launch(void* const* d_in, const int* in_sizes, int n_in,
                              void* d_out, int out_size, void* d_ws, size_t ws_size,
                              hipStream_t stream) {
    (void)in_sizes; (void)n_in; (void)out_size; (void)ws_size;
    const float* z  = (const float*)d_in[0];
    const float* cb = (const float*)d_in[1];
    float* out = (float*)d_out;
    char* ws = (char*)d_ws;

    float*     sums   = (float*)(ws + WS_SUMS);
    int*       hist   = (int*)(ws + WS_HIST);
    int*       idxbuf = (int*)(ws + WS_IDX);
    float*     cnorm  = (float*)(ws + WS_CNORM);
    float*     pval   = (float*)(ws + WS_PVAL);
    int*       pidx   = (int*)(ws + WS_PIDX);
    _Float16*  zh     = (_Float16*)(ws + WS_ZH);
    _Float16*  cbh    = (_Float16*)(ws + WS_CBH);

    vq_init_kernel<<<36, 256, 0, stream>>>((int*)ws);
    vq_prepz_kernel<<<4096, 256, 0, stream>>>(z, zh);
    vq_prepcb_kernel<<<N_E, 64, 0, stream>>>(cb, cbh, cnorm);
    vq_argmin_kernel<<<(NROWS / 64) * NCHUNK, 128, 0, stream>>>(
        zh, cbh, cnorm, pval, pidx);
    vq_reduce_kernel<<<NROWS / 256, 256, 0, stream>>>(pval, pidx, idxbuf);
    vq_gather_kernel<<<NELEM / 256, 256, 0, stream>>>(
        z, cb, idxbuf, out + OUT_ZQ, sums, hist, out + OUT_IDX);
    vq_onehot_kernel<<<4096, 256, 0, stream>>>(idxbuf, out + OUT_ME);
    vq_final_kernel<<<1, 256, 0, stream>>>(sums, hist, out);
}
